// SelfAttention3d_61701500174669
// MI455X (gfx1250) — matrix operand order repaired and checked by reading, not compiled, not run on hardware
//
#include <hip/hip_runtime.h>
#include <hip/hip_bf16.h>
#include <stdint.h>

// ---------------------------------------------------------------------------
// SelfAttention3d for MI455X (gfx1250): flash-attention fused with
// v_wmma_f32_16x16x32_bf16, wave32 layouts per CDNA5 ISA 7.12.2.
// B=2, C=64, Cr=8, N=16*16*16=4096.
// Round 3: DPP ctrl as template constant (builtin requires immediate).
// ---------------------------------------------------------------------------

#define C_CH  64
#define CR    8
#define NPOS  4096
#define BATCH 2

typedef __attribute__((ext_vector_type(16))) __bf16 v16bf;
typedef __attribute__((ext_vector_type(8)))  float  v8f;

union BF16x16 {
    v16bf        v;
    unsigned int u[8];
};

static __device__ __forceinline__ unsigned short f32_to_bf16(float f) {
    unsigned int u = __float_as_uint(f);
    u += 0x7FFFu + ((u >> 16) & 1u);   // round-to-nearest-even
    return (unsigned short)(u >> 16);
}

static __device__ __forceinline__ unsigned int pack_bf16(float lo, float hi) {
    return (unsigned int)f32_to_bf16(lo) | ((unsigned int)f32_to_bf16(hi) << 16);
}

// 16-lane butterfly reductions via DPP row ops (wave32: each half is one DPP
// row of 16, matching the WMMA C/D half-layout). Pure VALU, co-executes with
// the XDL WMMA pipeline; no LDS traffic.
#define DPP_QP_XOR1     0xB1   // quad_perm(1,0,3,2)
#define DPP_QP_XOR2     0x4E   // quad_perm(2,3,0,1)
#define DPP_HALF_MIRROR 0x141
#define DPP_MIRROR      0x140

template <int CTRL>
static __device__ __forceinline__ float dpp_f(float v) {
    return __int_as_float(__builtin_amdgcn_update_dpp(
        0, __float_as_int(v), CTRL, 0xF, 0xF, true));
}
static __device__ __forceinline__ float row16_max(float v) {
    v = fmaxf(v, dpp_f<DPP_QP_XOR1>(v));
    v = fmaxf(v, dpp_f<DPP_QP_XOR2>(v));
    v = fmaxf(v, dpp_f<DPP_HALF_MIRROR>(v));
    v = fmaxf(v, dpp_f<DPP_MIRROR>(v));
    return v;
}
static __device__ __forceinline__ float row16_sum(float v) {
    v += dpp_f<DPP_QP_XOR1>(v);
    v += dpp_f<DPP_QP_XOR2>(v);
    v += dpp_f<DPP_HALF_MIRROR>(v);
    v += dpp_f<DPP_MIRROR>(v);
    return v;
}

// ---------------------------------------------------------------------------
// Kernel 1: QKV projections (1x1 conv == channel GEMM). Memory-trivial
// (~0.08 GFLOP); plain VALU, outputs stored bf16 pre-arranged for WMMA:
//   qT,kT : [B][N][8]  (position-major, c contiguous -> one b128 per lane)
//   vB    : [B][C][N]  (channel-major, m contiguous  -> b128 B-tiles)
// ---------------------------------------------------------------------------
__global__ __launch_bounds__(256) void qkv_kernel(
    const float* __restrict__ x,
    const float* __restrict__ wq, const float* __restrict__ bq,
    const float* __restrict__ wk, const float* __restrict__ bk,
    const float* __restrict__ wv, const float* __restrict__ bv,
    unsigned short* __restrict__ qT,
    unsigned short* __restrict__ kT,
    unsigned short* __restrict__ vB)
{
    __shared__ float sWq[CR * C_CH], sWk[CR * C_CH], sWv[C_CH * C_CH];
    __shared__ float sBq[CR], sBk[CR], sBv[C_CH];

    for (int i = threadIdx.x; i < CR * C_CH; i += blockDim.x) {
        sWq[i] = wq[i];
        sWk[i] = wk[i];
    }
    for (int i = threadIdx.x; i < C_CH * C_CH; i += blockDim.x) sWv[i] = wv[i];
    if (threadIdx.x < CR) {
        sBq[threadIdx.x] = bq[threadIdx.x];
        sBk[threadIdx.x] = bk[threadIdx.x];
    }
    if (threadIdx.x < C_CH) sBv[threadIdx.x] = bv[threadIdx.x];
    __syncthreads();

    const int gid = blockIdx.x * blockDim.x + threadIdx.x;   // 0..8191
    const int b   = gid >> 12;
    const int n   = gid & (NPOS - 1);

    float xr[C_CH];
#pragma unroll 8
    for (int c = 0; c < C_CH; ++c)
        xr[c] = x[((size_t)(b * C_CH + c) << 12) + n];

    float qf[CR], kf[CR];
#pragma unroll
    for (int o = 0; o < CR; ++o) {
        float aq = sBq[o], ak = sBk[o];
#pragma unroll
        for (int c = 0; c < C_CH; ++c) {
            aq = fmaf(sWq[o * C_CH + c], xr[c], aq);
            ak = fmaf(sWk[o * C_CH + c], xr[c], ak);
        }
        qf[o] = aq;
        kf[o] = ak;
    }
    uint4 qv, kv;
    qv.x = pack_bf16(qf[0], qf[1]); qv.y = pack_bf16(qf[2], qf[3]);
    qv.z = pack_bf16(qf[4], qf[5]); qv.w = pack_bf16(qf[6], qf[7]);
    kv.x = pack_bf16(kf[0], kf[1]); kv.y = pack_bf16(kf[2], kf[3]);
    kv.z = pack_bf16(kf[4], kf[5]); kv.w = pack_bf16(kf[6], kf[7]);
    *(uint4*)(qT + ((size_t)gid << 3)) = qv;
    *(uint4*)(kT + ((size_t)gid << 3)) = kv;

#pragma unroll 4
    for (int o = 0; o < C_CH; ++o) {
        float a = sBv[o];
#pragma unroll
        for (int c = 0; c < C_CH; ++c) a = fmaf(sWv[o * C_CH + c], xr[c], a);
        vB[((size_t)(b * C_CH + o) << 12) + n] = f32_to_bf16(a);
    }
}

// ---------------------------------------------------------------------------
// Kernel 2: fused flash attention. One wave owns TWO 16-row query tiles so
// every K/V B-matrix load feeds 2 WMMAs (12 WMMAs per 32-wide m-chunk).
// Online softmax: per-iteration row-max via DPP butterflies; row-sum kept as
// per-lane partials (rescaled like the accumulators) and reduced once at the
// end. P goes through a per-wave LDS tile for the D->A layout transpose;
// wave-local s_wait_dscnt replaces block barriers (LDS is in-order per wave).
// ---------------------------------------------------------------------------
__global__ __launch_bounds__(256) void attn_kernel(
    const unsigned short* __restrict__ qT,
    const unsigned short* __restrict__ kT,
    const unsigned short* __restrict__ vB,
    float* __restrict__ avT)
{
    __shared__ unsigned short Psh[8][2 * 16 * 32];   // per-wave 2x(16x32) bf16

    const int wave = threadIdx.x >> 5;
    const int lane = threadIdx.x & 31;
    const int half = lane >> 4;      // 0: rows 0-7 / K 0-7,16-23 ; 1: rows 8-15 / K 8-15,24-31
    const int lx   = lane & 15;

    const int pair = blockIdx.x * 8 + wave;      // 0..255
    const int b    = pair >> 7;
    const int n0   = (pair & 127) << 5;          // 32 query rows per wave

    // A-matrices: q^T [16 rows n][K=32 c], c=0..7 nonzero (lanes 0-15, VGPR 0-3)
    BF16x16 aqA, aqB;
#pragma unroll
    for (int i = 0; i < 8; ++i) { aqA.u[i] = 0u; aqB.u[i] = 0u; }
    if (half == 0) {
        const uint4 tA = *(const uint4*)(qT + ((size_t)(b * NPOS + n0 + lx) << 3));
        const uint4 tB = *(const uint4*)(qT + ((size_t)(b * NPOS + n0 + 16 + lx) << 3));
        aqA.u[0] = tA.x; aqA.u[1] = tA.y; aqA.u[2] = tA.z; aqA.u[3] = tA.w;
        aqB.u[0] = tB.x; aqB.u[1] = tB.y; aqB.u[2] = tB.z; aqB.u[3] = tB.w;
    }

    v8f accA0 = {0, 0, 0, 0, 0, 0, 0, 0};
    v8f accA1 = accA0, accA2 = accA0, accA3 = accA0;
    v8f accB0 = accA0, accB1 = accA0, accB2 = accA0, accB3 = accA0;
    float rmA[8], rmB[8], lsA[8], lsB[8];
#pragma unroll
    for (int r = 0; r < 8; ++r) {
        rmA[r] = -1e30f; rmB[r] = -1e30f; lsA[r] = 0.0f; lsB[r] = 0.0f;
    }

    const unsigned short* kbase = kT + ((size_t)(b * NPOS) << 3);
    const unsigned short* vbase = vB + (size_t)(b * C_CH) * NPOS;
    unsigned short* pwA = &Psh[wave][0];
    unsigned short* pwB = &Psh[wave][512];

    for (int m0 = 0; m0 < NPOS; m0 += 32) {
        // ---- K B-matrix tiles [K=32 c][N=16 m]: c pairs contiguous per lane
        BF16x16 bk0, bk1;
#pragma unroll
        for (int i = 0; i < 8; ++i) { bk0.u[i] = 0u; bk1.u[i] = 0u; }
        if (half == 0) {
            uint4 t0 = *(const uint4*)(kbase + ((size_t)(m0 + lx) << 3));
            uint4 t1 = *(const uint4*)(kbase + ((size_t)(m0 + 16 + lx) << 3));
            bk0.u[0] = t0.x; bk0.u[1] = t0.y; bk0.u[2] = t0.z; bk0.u[3] = t0.w;
            bk1.u[0] = t1.x; bk1.u[1] = t1.y; bk1.u[2] = t1.z; bk1.u[3] = t1.w;
        }
        if (m0 + 32 < NPOS) {   // pull next K/V chunk toward WGP (global_prefetch_b8)
            __builtin_prefetch(kbase + ((size_t)(m0 + 32 + lx) << 3), 0, 1);
            __builtin_prefetch(vbase + (size_t)lx * NPOS + (m0 + 32), 0, 1);
        }

        v8f z = {0, 0, 0, 0, 0, 0, 0, 0};
        v8f sA0 = __builtin_amdgcn_wmma_f32_16x16x32_bf16(false, aqA.v, false, bk0.v,
                                                          (short)0, z, false, false);
        v8f sA1 = __builtin_amdgcn_wmma_f32_16x16x32_bf16(false, aqA.v, false, bk1.v,
                                                          (short)0, z, false, false);
        v8f sB0 = __builtin_amdgcn_wmma_f32_16x16x32_bf16(false, aqB.v, false, bk0.v,
                                                          (short)0, z, false, false);
        v8f sB1 = __builtin_amdgcn_wmma_f32_16x16x32_bf16(false, aqB.v, false, bk1.v,
                                                          (short)0, z, false, false);

        // ---- online softmax, tile A (row-max via DPP; l kept per-lane)
#pragma unroll
        for (int r = 0; r < 8; ++r) {
            float tm = row16_max(fmaxf(sA0[r], sA1[r]));
            float nm = fmaxf(rmA[r], tm);
            float sc = __expf(rmA[r] - nm);
            rmA[r] = nm;
            float p0 = __expf(sA0[r] - nm);
            float p1 = __expf(sA1[r] - nm);
            lsA[r] = lsA[r] * sc + p0 + p1;
            accA0[r] *= sc; accA1[r] *= sc; accA2[r] *= sc; accA3[r] *= sc;
            const int row = half * 8 + r;
            pwA[row * 32 + lx]      = f32_to_bf16(p0);
            pwA[row * 32 + 16 + lx] = f32_to_bf16(p1);
        }
        // ---- online softmax, tile B
#pragma unroll
        for (int r = 0; r < 8; ++r) {
            float tm = row16_max(fmaxf(sB0[r], sB1[r]));
            float nm = fmaxf(rmB[r], tm);
            float sc = __expf(rmB[r] - nm);
            rmB[r] = nm;
            float p0 = __expf(sB0[r] - nm);
            float p1 = __expf(sB1[r] - nm);
            lsB[r] = lsB[r] * sc + p0 + p1;
            accB0[r] *= sc; accB1[r] *= sc; accB2[r] *= sc; accB3[r] *= sc;
            const int row = half * 8 + r;
            pwB[row * 32 + lx]      = f32_to_bf16(p0);
            pwB[row * 32 + 16 + lx] = f32_to_bf16(p1);
        }

        // Wave-local RAW on the per-wave P tile: LDS is in-order per wave, so
        // a dscnt drain (no block barrier) is sufficient.
        asm volatile("s_wait_dscnt 0" ::: "memory");

        BF16x16 a2A, a2B;   // lane: row=lx, K = half*8+{0..7} and 16+half*8+{0..7}
        {
            uint4 t0 = *(const uint4*)(pwA + lx * 32 + half * 8);
            uint4 t1 = *(const uint4*)(pwA + lx * 32 + 16 + half * 8);
            a2A.u[0] = t0.x; a2A.u[1] = t0.y; a2A.u[2] = t0.z; a2A.u[3] = t0.w;
            a2A.u[4] = t1.x; a2A.u[5] = t1.y; a2A.u[6] = t1.z; a2A.u[7] = t1.w;
            uint4 t2 = *(const uint4*)(pwB + lx * 32 + half * 8);
            uint4 t3 = *(const uint4*)(pwB + lx * 32 + 16 + half * 8);
            a2B.u[0] = t2.x; a2B.u[1] = t2.y; a2B.u[2] = t2.z; a2B.u[3] = t2.w;
            a2B.u[4] = t3.x; a2B.u[5] = t3.y; a2B.u[6] = t3.z; a2B.u[7] = t3.w;
        }

        // ---- V B-matrices [K=32 m][N=16 c], each feeding both query tiles
#define AV_TILE(ACCA, ACCB, T)                                                  \
        {                                                                       \
            const unsigned short* vp =                                          \
                vbase + (size_t)((T) * 16 + lx) * NPOS + m0 + half * 8;         \
            uint4 w0 = *(const uint4*)vp;                                       \
            uint4 w1 = *(const uint4*)(vp + 16);                                \
            BF16x16 b2;                                                         \
            b2.u[0] = w0.x; b2.u[1] = w0.y; b2.u[2] = w0.z; b2.u[3] = w0.w;     \
            b2.u[4] = w1.x; b2.u[5] = w1.y; b2.u[6] = w1.z; b2.u[7] = w1.w;     \
            ACCA = __builtin_amdgcn_wmma_f32_16x16x32_bf16(false, a2A.v, false, \
                       b2.v, (short)0, ACCA, false, false);                     \
            ACCB = __builtin_amdgcn_wmma_f32_16x16x32_bf16(false, a2B.v, false, \
                       b2.v, (short)0, ACCB, false, false);                     \
        }
        AV_TILE(accA0, accB0, 0)
        AV_TILE(accA1, accB1, 1)
        AV_TILE(accA2, accB2, 2)
        AV_TILE(accA3, accB3, 3)
#undef AV_TILE

        // Keep next iteration's P stores from hoisting above this iteration's
        // LDS reads (WAR); LDS executes in wave issue order.
        asm volatile("" ::: "memory");
    }

    // ---- finalize: reduce deferred row sums once, divide, write avT f32
    float* outA = avT + ((size_t)(b * NPOS + n0 + half * 8) << 6);
    float* outB = avT + ((size_t)(b * NPOS + n0 + 16 + half * 8) << 6);
#pragma unroll
    for (int r = 0; r < 8; ++r) {
        const float invA = 1.0f / row16_sum(lsA[r]);
        const float invB = 1.0f / row16_sum(lsB[r]);
        float* oA = outA + ((size_t)r << 6) + lx;
        float* oB = outB + ((size_t)r << 6) + lx;
        oA[0]  = accA0[r] * invA;
        oA[16] = accA1[r] * invA;
        oA[32] = accA2[r] * invA;
        oA[48] = accA3[r] * invA;
        oB[0]  = accB0[r] * invB;
        oB[16] = accB1[r] * invB;
        oB[32] = accB2[r] * invB;
        oB[48] = accB3[r] * invB;
    }
}

// ---------------------------------------------------------------------------
// Kernel 3: output projection y = wa @ av + ba, plus per-channel BN partial
// sums (LDS reduce -> global f32 atomics).
// ---------------------------------------------------------------------------
__global__ __launch_bounds__(256) void proj_kernel(
    const float* __restrict__ avT, const float* __restrict__ wa,
    const float* __restrict__ ba, float* __restrict__ y,
    float* __restrict__ gsum, float* __restrict__ gsumsq)
{
    __shared__ float sW[C_CH * C_CH];
    __shared__ float sSum[C_CH], sSq[C_CH];

    for (int i = threadIdx.x; i < C_CH * C_CH; i += blockDim.x) sW[i] = wa[i];
    if (threadIdx.x < C_CH) { sSum[threadIdx.x] = 0.f; sSq[threadIdx.x] = 0.f; }
    __syncthreads();

    const int gid = blockIdx.x * blockDim.x + threadIdx.x;   // 0..8191
    const int b   = gid >> 12;
    const int n   = gid & (NPOS - 1);

    float a[C_CH];
    const float4* ap = (const float4*)(avT + ((size_t)gid << 6));
#pragma unroll
    for (int i = 0; i < 16; ++i) {
        float4 t = ap[i];
        a[4 * i] = t.x; a[4 * i + 1] = t.y; a[4 * i + 2] = t.z; a[4 * i + 3] = t.w;
    }

#pragma unroll 4
    for (int o = 0; o < C_CH; ++o) {
        float acc = ba[o];
#pragma unroll
        for (int c = 0; c < C_CH; ++c) acc = fmaf(sW[o * C_CH + c], a[c], acc);
        y[((size_t)(b * C_CH + o) << 12) + n] = acc;
        atomicAdd(&sSum[o], acc);
        atomicAdd(&sSq[o], acc * acc);
    }
    __syncthreads();
    if (threadIdx.x < C_CH) {
        atomicAdd(&gsum[threadIdx.x],   sSum[threadIdx.x]);
        atomicAdd(&gsumsq[threadIdx.x], sSq[threadIdx.x]);
    }
}

// ---------------------------------------------------------------------------
// Kernel 4: BN (training stats over B*N=8192), gamma/beta, relu, residual.
// ---------------------------------------------------------------------------
__global__ __launch_bounds__(256) void bn_finalize_kernel(
    const float* __restrict__ y, const float* __restrict__ x,
    const float* __restrict__ gsum, const float* __restrict__ gsumsq,
    const float* __restrict__ gamma, const float* __restrict__ beta,
    float* __restrict__ out)
{
    const int idx = blockIdx.x * blockDim.x + threadIdx.x;
    if (idx >= BATCH * C_CH * NPOS) return;
    const int c = (idx >> 12) & (C_CH - 1);

    const float inv_n = 1.0f / (float)(BATCH * NPOS);
    const float mean  = gsum[c] * inv_n;
    const float var   = gsumsq[c] * inv_n - mean * mean;
    float v = (y[idx] - mean) * rsqrtf(var + 1e-5f);
    v = v * gamma[c] + beta[c];
    v = fmaxf(v, 0.0f);
    out[idx] = v + x[idx];
}

// ---------------------------------------------------------------------------
extern "C" void kernel_launch(void* const* d_in, const int* in_sizes, int n_in,
                              void* d_out, int out_size, void* d_ws, size_t ws_size,
                              hipStream_t stream)
{
    const float* x     = (const float*)d_in[0];
    const float* wq    = (const float*)d_in[1];
    const float* bq    = (const float*)d_in[2];
    const float* wk    = (const float*)d_in[3];
    const float* bk    = (const float*)d_in[4];
    const float* wv    = (const float*)d_in[5];
    const float* bv    = (const float*)d_in[6];
    const float* wa    = (const float*)d_in[7];
    const float* ba    = (const float*)d_in[8];
    const float* gamma = (const float*)d_in[9];
    const float* beta  = (const float*)d_in[10];

    char* ws = (char*)d_ws;
    unsigned short* qT  = (unsigned short*)(ws + 0);         // 128 KB
    unsigned short* kT  = (unsigned short*)(ws + 131072);    // 128 KB
    unsigned short* vB  = (unsigned short*)(ws + 262144);    // 1 MB
    float*          avT = (float*)(ws + 1310720);            // 2 MB
    float*          yB  = (float*)(ws + 3407872);            // 2 MB
    float*          st  = (float*)(ws + 5505024);            // 512 B (sum, sumsq)

    (void)hipMemsetAsync(st, 0, 2 * C_CH * sizeof(float), stream);

    qkv_kernel<<<(BATCH * NPOS) / 256, 256, 0, stream>>>(
        x, wq, bq, wk, bk, wv, bv, qT, kT, vB);

    // 256 query-tile pairs (32 rows each), 8 waves per block
    attn_kernel<<<32, 256, 0, stream>>>(qT, kT, vB, avT);

    proj_kernel<<<(BATCH * NPOS) / 256, 256, 0, stream>>>(
        avT, wa, ba, yB, st, st + C_CH);

    bn_finalize_kernel<<<(BATCH * C_CH * NPOS) / 256, 256, 0, stream>>>(
        yB, x, st, st + C_CH, gamma, beta, (float*)d_out);
}
